// GATv2_Model_48112223649930
// MI455X (gfx1250) — matrix operand order repaired and checked by reading, not compiled, not run on hardware
//
#include <hip/hip_runtime.h>

// ---------------------------------------------------------------------------
// GATv2 forward for MI455X (gfx1250, wave32, WMMA).
// GEMMs: v_wmma_f32_16x16x32_f16, A tile staged into LDS with
// global_load_async_to_lds_b128 (ASYNCcnt path). Edge phase is L2-resident
// gather/scatter with float atomics. All launches on `stream`.
// ---------------------------------------------------------------------------

typedef __attribute__((ext_vector_type(16))) _Float16 v16h;
typedef __attribute__((ext_vector_type(8)))  _Float16 v8h;
typedef __attribute__((ext_vector_type(8)))  float    v8f;

#define HEADS 4
#define KDIM  128   // input dim of every layer (IN = H*F = 128)
#define SLOPE 0.2f
#define LDS_STRIDE 136  // 128 halves + 8 pad halves (272B = 68 banks: conflict-free)

// ---------------------------------------------------------------- helpers ---
__device__ __forceinline__ void atomicMaxFloat(float* addr, float val) {
  // sign-split trick: monotone over float ordering, always compiles
  if (val >= 0.0f) atomicMax((int*)addr, __float_as_int(val));
  else             atomicMin((unsigned int*)addr, __float_as_uint(val));
}

__device__ __forceinline__ v16h combine16(v8h lo, v8h hi) {
  return __builtin_shufflevector(lo, hi, 0, 1, 2, 3, 4, 5, 6, 7,
                                 8, 9, 10, 11, 12, 13, 14, 15);
}

// ------------------------------------------------------------ conversions ---
__global__ void f32_to_f16_kernel(const float* __restrict__ in,
                                  _Float16* __restrict__ out, int n) {
  int i = blockIdx.x * blockDim.x + threadIdx.x;
  if (i < n) out[i] = (_Float16)in[i];
}

__global__ void init_softmax_bufs(float* __restrict__ maxbuf,
                                  float* __restrict__ zbuf, int n) {
  int i = blockIdx.x * blockDim.x + threadIdx.x;
  if (i < n) { maxbuf[i] = -3.0e38f; zbuf[i] = 0.0f; }
}

// ---------------------------------------------------------------- WMMA GEMM -
// C[M, Fout] = A[M, 128] * W[Fout, 128]^T + bias.
// Block = 8 waves; one 16-row M tile per block (A staged once in LDS, shared
// by all 8 waves), one 16-col N tile per wave.
__global__ __launch_bounds__(256)
void wmma_gemm_bias(const _Float16* __restrict__ A, const _Float16* __restrict__ W,
                    const float* __restrict__ bias, float* __restrict__ C,
                    int M, int Fout, int ntiles) {
  __shared__ _Float16 atile[16 * LDS_STRIDE];

  const int tid   = threadIdx.x;
  const int wave  = tid >> 5;
  const int lane  = tid & 31;
  const int mtile = blockIdx.x;

  // ---- stage 16x128 f16 A tile into LDS via async copy (16B per thread) ----
  {
    const int r = tid >> 4;          // 0..15 : tile row
    const int c = (tid & 15) * 8;    // 0..120: half-column
    int gr = mtile * 16 + r;
    if (gr >= M) gr = M - 1;         // clamp (M=50000 is an exact multiple of 16)
    const _Float16* gp = A + (size_t)gr * KDIM + c;
    const unsigned lds_off = (unsigned)(size_t)(void*)&atile[r * LDS_STRIDE + c];
    asm volatile("global_load_async_to_lds_b128 %0, %1, off"
                 :: "v"(lds_off), "v"(gp) : "memory");
    asm volatile("s_wait_asynccnt 0x0" ::: "memory");
  }
  __syncthreads();

  const int ntile = blockIdx.y * 8 + wave;
  if (ntile >= ntiles) return;

  const int row  = lane & 15;   // A row / B column held by this lane
  const int hsel = lane >> 4;   // lane-half select

  const int wrow = ntile * 16 + row;  // output-feature index (B column)
  const _Float16* brow = W + (size_t)(wrow < Fout ? wrow : Fout - 1) * KDIM;
  const _Float16* asrc = &atile[row * LDS_STRIDE];

  v8f acc = {};
#pragma unroll
  for (int kb = 0; kb < KDIM; kb += 32) {
    // A 16x32 f16 fragment: lanes 0-15 hold K={kb..kb+7, kb+16..kb+23},
    // lanes 16-31 hold the +8 variants (ISA §7.12.2 16-bit A table).
    const int ka = kb + hsel * 8;
    v16h a = combine16(*(const v8h*)(asrc + ka),
                       *(const v8h*)(asrc + ka + 16));
    // B 32x16 f16 fragment: same K order as A: lanes 0-15 hold
    // K={kb..kb+7, kb+16..kb+23}, lanes 16-31 hold the +8 variants.
    const int kq = kb + hsel * 8;
    v16h b = combine16(*(const v8h*)(brow + kq),
                       *(const v8h*)(brow + kq + 16));
    acc = __builtin_amdgcn_wmma_f32_16x16x32_f16(
        /*neg_a=*/false, a, /*neg_b=*/false, b,
        /*c_mod=*/(short)0, acc, /*reuse_a=*/false, /*reuse_b=*/false);
  }

  const int col = ntile * 16 + (lane & 15);
  if (col >= Fout) return;
  const float bv = bias[col];
  float* cp = C + (size_t)(mtile * 16 + hsel * 8) * Fout + col;
  if (mtile * 16 + 16 <= M) {  // uniform full-tile fast path (no EXEC churn)
#pragma unroll
    for (int r = 0; r < 8; ++r) cp[(size_t)r * Fout] = acc[r] + bv;
  } else {
    for (int r = 0; r < 8; ++r)
      if (mtile * 16 + hsel * 8 + r < M) cp[(size_t)r * Fout] = acc[r] + bv;
  }
}

// ------------------------------------------------------------- edge phase ---
// one wave per edge: score[e][h] = attn[h] . leaky_relu(fs[src] + fd[dst])
__global__ __launch_bounds__(256)
void edge_score_kernel(const float* __restrict__ fs, const float* __restrict__ fd,
                       const float* __restrict__ attn,
                       const int* __restrict__ src, const int* __restrict__ dst,
                       float* __restrict__ sbuf, float* __restrict__ maxbuf,
                       int E, int fdim, int stride) {
  const int e = blockIdx.x * (blockDim.x >> 5) + (threadIdx.x >> 5);
  if (e >= E) return;
  const int lane = threadIdx.x & 31;
  const int s = src[e], d = dst[e];
  const float* ps = fs + (size_t)s * stride;
  const float* pd = fd + (size_t)d * stride;
#pragma unroll
  for (int h = 0; h < HEADS; ++h) {
    float acc = 0.0f;
    for (int f = lane; f < fdim; f += 32) {
      float v = ps[h * fdim + f] + pd[h * fdim + f];
      v = v > 0.0f ? v : SLOPE * v;
      acc += v * attn[h * fdim + f];
    }
#pragma unroll
    for (int o = 16; o > 0; o >>= 1) acc += __shfl_xor(acc, o, 32);
    if (lane == 0) {
      sbuf[(size_t)e * HEADS + h] = acc;
      atomicMaxFloat(&maxbuf[d * HEADS + h], acc);
    }
  }
}

// one thread per (edge, head): s = exp(score - max[dst]); z[dst] += s
__global__ void edge_exp_kernel(const int* __restrict__ dst,
                                float* __restrict__ sbuf,
                                const float* __restrict__ maxbuf,
                                float* __restrict__ zbuf, int EH) {
  const int i = blockIdx.x * blockDim.x + threadIdx.x;
  if (i >= EH) return;
  const int e = i >> 2, h = i & 3;
  const int d = dst[e];
  const float sv = __expf(sbuf[i] - maxbuf[d * HEADS + h]);
  sbuf[i] = sv;
  atomicAdd(&zbuf[d * HEADS + h], sv);
}

// one wave per edge: rst[dst] += fs[src] * (s / z[dst])
__global__ __launch_bounds__(256)
void edge_scatter_kernel(const float* __restrict__ fs,
                         const int* __restrict__ src, const int* __restrict__ dst,
                         const float* __restrict__ sbuf, const float* __restrict__ zbuf,
                         float* __restrict__ rst, int E, int fdim, int stride) {
  const int e = blockIdx.x * (blockDim.x >> 5) + (threadIdx.x >> 5);
  if (e >= E) return;
  const int lane = threadIdx.x & 31;
  const int s = src[e], d = dst[e];
  const float* ps = fs + (size_t)s * stride;
  float* pr = rst + (size_t)d * stride;
#pragma unroll
  for (int h = 0; h < HEADS; ++h) {
    const float a = sbuf[(size_t)e * HEADS + h] / zbuf[d * HEADS + h];
    for (int f = lane; f < fdim; f += 32)
      atomicAdd(&pr[h * fdim + f], ps[h * fdim + f] * a);
  }
}

// ------------------------------------------------ head-mean + log-softmax ---
__global__ __launch_bounds__(256)
void mean_logsoftmax_kernel(const float* __restrict__ rst,
                            float* __restrict__ out, int N) {
  const int node = blockIdx.x * (blockDim.x >> 5) + (threadIdx.x >> 5);
  if (node >= N) return;
  const int lane = threadIdx.x & 31;
  const int NC = 47, stride = HEADS * NC;  // 188
  const float* p = rst + (size_t)node * stride;
  const int c0 = lane, c1 = lane + 32;
  float v0 = -3.0e38f, v1 = -3.0e38f;
  if (c0 < NC) {
    float s = 0.0f;
#pragma unroll
    for (int h = 0; h < HEADS; ++h) s += p[h * NC + c0];
    v0 = 0.25f * s;
  }
  if (c1 < NC) {
    float s = 0.0f;
#pragma unroll
    for (int h = 0; h < HEADS; ++h) s += p[h * NC + c1];
    v1 = 0.25f * s;
  }
  float mx = fmaxf(v0, v1);
#pragma unroll
  for (int o = 16; o > 0; o >>= 1) mx = fmaxf(mx, __shfl_xor(mx, o, 32));
  float sum = (c0 < NC ? __expf(v0 - mx) : 0.0f) + (c1 < NC ? __expf(v1 - mx) : 0.0f);
#pragma unroll
  for (int o = 16; o > 0; o >>= 1) sum += __shfl_xor(sum, o, 32);
  const float lse = logf(sum);
  if (c0 < NC) out[(size_t)node * NC + c0] = v0 - mx - lse;
  if (c1 < NC) out[(size_t)node * NC + c1] = v1 - mx - lse;
}

// ------------------------------------------------------------------- host ---
static inline int ceil_div(int a, int b) { return (a + b - 1) / b; }

extern "C" void kernel_launch(void* const* d_in, const int* in_sizes, int n_in,
                              void* d_out, int out_size, void* d_ws, size_t ws_size,
                              hipStream_t stream) {
  (void)n_in; (void)out_size; (void)ws_size;
  const float* x   = (const float*)d_in[0];
  const int*   src = (const int*)d_in[1];
  const int*   dst = (const int*)d_in[2];
  const int N = in_sizes[0] / KDIM;   // 50000
  const int E = in_sizes[1];          // 800000

  struct Layer {
    const float *Ws, *bs, *Wd, *bd, *attn, *Wr, *br;
    int Fout, fdim;
  } L[3];
  for (int l = 0; l < 3; ++l) {
    const int b = 3 + 7 * l;  // params flattened in dict insertion order
    L[l].Ws = (const float*)d_in[b + 0];
    L[l].bs = (const float*)d_in[b + 1];
    L[l].Wd = (const float*)d_in[b + 2];
    L[l].bd = (const float*)d_in[b + 3];
    L[l].attn = (const float*)d_in[b + 4];
    L[l].Wr = (const float*)d_in[b + 5];
    L[l].br = (const float*)d_in[b + 6];
  }
  L[0].Fout = 128; L[0].fdim = 32;
  L[1].Fout = 128; L[1].fdim = 32;
  L[2].Fout = HEADS * 47; L[2].fdim = 47;   // 188

  // ---- carve workspace (256B aligned) ----
  char* base = (char*)d_ws;
  size_t off = 0;
  auto carve = [&](size_t bytes) -> void* {
    off = (off + 255) & ~(size_t)255;
    void* p = base + off;
    off += bytes;
    return p;
  };
  const size_t maxFeat = (size_t)N * 188;
  _Float16* h16    = (_Float16*)carve((size_t)N * KDIM * sizeof(_Float16));
  _Float16* w16    = (_Float16*)carve((size_t)192 * KDIM * sizeof(_Float16));
  float*    fs     = (float*)carve(maxFeat * sizeof(float));
  float*    fd     = (float*)carve(maxFeat * sizeof(float));
  float*    rst    = (float*)carve(maxFeat * sizeof(float));
  float*    sbuf   = (float*)carve((size_t)E * HEADS * sizeof(float));
  float*    maxbuf = (float*)carve((size_t)N * HEADS * sizeof(float));
  float*    zbuf   = (float*)carve((size_t)N * HEADS * sizeof(float));

  const dim3 blk256(256);
  const int mtiles = ceil_div(N, 16);

  const float* h_in = x;  // f32 input for layer 0; rst afterwards
  for (int l = 0; l < 3; ++l) {
    const int Fout = L[l].Fout, fdim = L[l].fdim;
    const int ntiles = ceil_div(Fout, 16);
    const dim3 gemm_grid(mtiles, ceil_div(ntiles, 8));

    // h -> f16 (layer input dim is always 128)
    {
      const int cnt = N * KDIM;
      f32_to_f16_kernel<<<ceil_div(cnt, 256), blk256, 0, stream>>>(h_in, h16, cnt);
    }
    // fs = h Ws^T + bs
    f32_to_f16_kernel<<<ceil_div(Fout * KDIM, 256), blk256, 0, stream>>>(L[l].Ws, w16, Fout * KDIM);
    wmma_gemm_bias<<<gemm_grid, blk256, 0, stream>>>(h16, w16, L[l].bs, fs, N, Fout, ntiles);
    // fd = h Wd^T + bd
    f32_to_f16_kernel<<<ceil_div(Fout * KDIM, 256), blk256, 0, stream>>>(L[l].Wd, w16, Fout * KDIM);
    wmma_gemm_bias<<<gemm_grid, blk256, 0, stream>>>(h16, w16, L[l].bd, fd, N, Fout, ntiles);
    // rst = res = h Wr^T + br   (scatter accumulates on top)
    f32_to_f16_kernel<<<ceil_div(Fout * KDIM, 256), blk256, 0, stream>>>(L[l].Wr, w16, Fout * KDIM);
    wmma_gemm_bias<<<gemm_grid, blk256, 0, stream>>>(h16, w16, L[l].br, rst, N, Fout, ntiles);

    // edge softmax + aggregation
    init_softmax_bufs<<<ceil_div(N * HEADS, 256), blk256, 0, stream>>>(maxbuf, zbuf, N * HEADS);
    edge_score_kernel<<<ceil_div(E, 8), blk256, 0, stream>>>(fs, fd, L[l].attn, src, dst,
                                                             sbuf, maxbuf, E, fdim, Fout);
    edge_exp_kernel<<<ceil_div(E * HEADS, 256), blk256, 0, stream>>>(dst, sbuf, maxbuf, zbuf,
                                                                     E * HEADS);
    edge_scatter_kernel<<<ceil_div(E, 8), blk256, 0, stream>>>(fs, src, dst, sbuf, zbuf,
                                                               rst, E, fdim, Fout);
    h_in = rst;
  }

  // mean over heads + log_softmax -> d_out [N, 47] f32
  mean_logsoftmax_kernel<<<ceil_div(N, 8), blk256, 0, stream>>>(rst, (float*)d_out, N);
}
